// Griffin_21380347199575
// MI455X (gfx1250) — matrix operand order repaired and checked by reading, not compiled, not run on hardware
//
#include <hip/hip_runtime.h>
#include <math.h>
#include <stdint.h>

// ---------------------------------------------------------------------------
// Griffin block for MI455X (gfx1250, wave32, WMMA).
// GEMMs: bf16 inputs, f32 accumulate, v_wmma_f32_16x16x32_bf16, with
// double-buffered GLOBAL_LOAD_ASYNC_TO_LDS_B128 tile staging (ASYNCcnt).
// Attention: WMMA flash-attention, 16-query tiles, online softmax.
// Hawk recurrence: chunked 3-pass parallel linear scan.
// ---------------------------------------------------------------------------

typedef __bf16 bf16_t;
typedef __attribute__((ext_vector_type(16))) __bf16 v16bf;
typedef __attribute__((ext_vector_type(8)))  __bf16 v8bf;
typedef __attribute__((ext_vector_type(8)))  float  v8f;

static constexpr int cB  = 2;
static constexpr int cT  = 4096;
static constexpr int cD  = 1024;
static constexpr int cHD = 128;
static constexpr int cHQ = 8;
static constexpr int cW  = 1024;
static constexpr int cHH = 1536;
static constexpr int cHG = 2048;
static constexpr int cK  = 4;          // conv taps
static constexpr int cM  = cB * cT;    // 8192 tokens
static constexpr int kSC  = 256;       // scan chunk length
static constexpr int kNCH = cT / kSC;  // 16 chunks

// ------------------------------ small helpers ------------------------------

__device__ inline float waveSum(float v) {
#pragma unroll
  for (int off = 16; off > 0; off >>= 1) v += __shfl_xor(v, off, 32);
  return v;
}

__device__ inline float gelu_exact(float x) {
  return 0.5f * x * (1.f + erff(x * 0.70710678118654752f));
}

// async 16B global -> LDS copy (per lane), tracked by ASYNCcnt
__device__ inline void async_copy16(unsigned lds_addr, unsigned long long gaddr) {
  asm volatile("global_load_async_to_lds_b128 %0, %1, off"
               :: "v"(lds_addr), "v"(gaddr) : "memory");
}
__device__ inline void async_wait0() {
  asm volatile("s_wait_asynccnt 0" ::: "memory");
}

// ------------------------------ elementwise --------------------------------

__global__ void k_f32_to_bf16(const float* __restrict__ s, bf16_t* __restrict__ d, int n) {
  int i = blockIdx.x * 256 + threadIdx.x;
  if (i < n) d[i] = (bf16_t)s[i];
}

__global__ void k_copy_f32(const float* __restrict__ s, float* __restrict__ d, int n) {
  int i = blockIdx.x * 256 + threadIdx.x;
  if (i < n) d[i] = s[i];
}

// RMSNorm over rows of width D, emitting bf16
__global__ void k_rmsnorm_bf16(const float* __restrict__ x, const float* __restrict__ g,
                               bf16_t* __restrict__ o, int D) {
  int row = blockIdx.x;
  const float* xr = x + (size_t)row * D;
  float ss = 0.f;
  for (int i = threadIdx.x; i < D; i += blockDim.x) { float v = xr[i]; ss += v * v; }
  ss = waveSum(ss);
  __shared__ float red[8];
  int wave = threadIdx.x >> 5, lane = threadIdx.x & 31;
  if (lane == 0) red[wave] = ss;
  __syncthreads();
  if (wave == 0) {
    float v = (lane < (int)(blockDim.x >> 5)) ? red[lane] : 0.f;
    v = waveSum(v);
    if (lane == 0) red[0] = v;
  }
  __syncthreads();
  float scale = sqrtf((float)D) * rsqrtf(red[0]);
  for (int i = threadIdx.x; i < D; i += blockDim.x)
    o[(size_t)row * D + i] = (bf16_t)(g[i] * scale * xr[i]);
}

// -------------------------- WMMA bf16 GEMM ---------------------------------
// C[M,N] = A[M,K] * W[N,K]^T (+bias) ; EPI==1 -> C += result (residual fuse)
// 256 threads / 8 waves; tile 64(M) x 128(N); K-step 32; double-buffered
// async-to-LDS staging.

template <int EPI>
__global__ __launch_bounds__(256)
void k_gemm_bf16_wmma(const bf16_t* __restrict__ A, const bf16_t* __restrict__ Wt,
                      float* __restrict__ C, const float* __restrict__ bias,
                      int M, int N, int K) {
  __shared__ __align__(16) __bf16 As[2][64][40];    // 80B rows, 16B aligned
  __shared__ __align__(16) __bf16 Bs[2][128][40];

  const int tid  = threadIdx.x;
  const int wave = tid >> 5;
  const int lane = tid & 31;
  const int wm = wave & 1;
  const int wn = wave >> 1;
  const int lane16   = lane & 15;
  const int laneHalf = lane >> 4;

  const int mTile = blockIdx.y * 64;
  const int nTile = blockIdx.x * 128;

  const int ldRowA = tid >> 2;
  const int ldColA = (tid & 3) * 8;

  auto stage = [&](int k0, int pbuf) {
    async_copy16((unsigned)(uintptr_t)&As[pbuf][ldRowA][ldColA],
                 (unsigned long long)(uintptr_t)(A + (size_t)(mTile + ldRowA) * K + k0 + ldColA));
#pragma unroll
    for (int c = 0; c < 2; ++c) {
      int t2  = tid + c * 256;
      int row = t2 >> 2;
      int col = (t2 & 3) * 8;
      async_copy16((unsigned)(uintptr_t)&Bs[pbuf][row][col],
                   (unsigned long long)(uintptr_t)(Wt + (size_t)(nTile + row) * K + k0 + col));
    }
  };

  v8f acc[2][2] = {};
  stage(0, 0);

  int it = 0;
  for (int k0 = 0; k0 < K; k0 += 32, ++it) {
    async_wait0();
    __syncthreads();
    const int cur = it & 1;
    if (k0 + 32 < K) stage(k0 + 32, cur ^ 1);

    v16bf afrag[2], bfrag[2];
#pragma unroll
    for (int i = 0; i < 2; ++i) {
      // A 16x32: half 0 -> K {0..7,16..23}; half 1 -> K {8..15,24..31}
      int ra = wm * 32 + i * 16 + lane16;
      v8bf a0 = *(const v8bf*)&As[cur][ra][8 * laneHalf];
      v8bf a1 = *(const v8bf*)&As[cur][ra][16 + 8 * laneHalf];
#pragma unroll
      for (int e = 0; e < 8; ++e) { afrag[i][e] = a0[e]; afrag[i][8 + e] = a1[e]; }
      // B 32x16: column = lane16, half 0 -> K {0..7,16..23}; half 1 -> K {8..15,24..31}
      int rb = wn * 32 + i * 16 + lane16;
      v8bf b0 = *(const v8bf*)&Bs[cur][rb][8 * laneHalf];
      v8bf b1 = *(const v8bf*)&Bs[cur][rb][16 + 8 * laneHalf];
#pragma unroll
      for (int e = 0; e < 8; ++e) { bfrag[i][e] = b0[e]; bfrag[i][8 + e] = b1[e]; }
    }

#pragma unroll
    for (int i = 0; i < 2; ++i)
#pragma unroll
      for (int j = 0; j < 2; ++j)
        acc[i][j] = __builtin_amdgcn_wmma_f32_16x16x32_bf16(
            false, afrag[i], false, bfrag[j], (short)0, acc[i][j], false, false);
    __syncthreads();
  }

#pragma unroll
  for (int i = 0; i < 2; ++i) {
#pragma unroll
    for (int j = 0; j < 2; ++j) {
      int n = nTile + wn * 32 + j * 16 + lane16;
      float bv = bias ? bias[n] : 0.f;
#pragma unroll
      for (int r = 0; r < 8; ++r) {
        int m = mTile + wm * 32 + i * 16 + r + (laneHalf ? 8 : 0);
        size_t off = (size_t)m * N + n;
        float v = acc[i][j][r] + bv;
        if (EPI == 0) C[off] = v; else C[off] += v;
      }
    }
  }
}

// ------------------------------- RoPE --------------------------------------
// Qf f32 [M,1024] -> Qb bf16 (roped); KVf [M,256] -> Kb bf16 [M,128] (roped),
// Vt bf16 [128, M] (transposed copy for WMMA B-fragments)
__global__ void k_rope(const float* __restrict__ qf, const float* __restrict__ kv,
                       bf16_t* __restrict__ qb, bf16_t* __restrict__ kb,
                       bf16_t* __restrict__ vt) {
  int row = blockIdx.x;
  int t   = row % cT;
  int tid = threadIdx.x;
  for (int pIdx = tid; pIdx < cHQ * cHD / 2; pIdx += blockDim.x) {
    int head = pIdx / (cHD / 2);
    int i    = pIdx % (cHD / 2);
    float inv = powf(10000.f, -2.f * (float)i / (float)cHD);
    float ang = (float)t * inv;
    float cs = cosf(ang), sn = sinf(ang);
    size_t base = (size_t)row * (cHQ * cHD) + head * cHD + 2 * i;
    float x1 = qf[base], x2 = qf[base + 1];
    qb[base]     = (bf16_t)(x1 * cs - x2 * sn);
    qb[base + 1] = (bf16_t)(x1 * sn + x2 * cs);
  }
  if (tid < cHD / 2) {
    int i = tid;
    float inv = powf(10000.f, -2.f * (float)i / (float)cHD);
    float ang = (float)t * inv;
    float cs = cosf(ang), sn = sinf(ang);
    size_t kbo = (size_t)row * 256;
    float x1 = kv[kbo + 2 * i], x2 = kv[kbo + 2 * i + 1];
    kb[(size_t)row * cHD + 2 * i]     = (bf16_t)(x1 * cs - x2 * sn);
    kb[(size_t)row * cHD + 2 * i + 1] = (bf16_t)(x1 * sn + x2 * cs);
  }
  for (int i = tid; i < cHD; i += blockDim.x)
    vt[(size_t)i * cM + row] = (bf16_t)kv[(size_t)row * 256 + cHD + i];
}

// --------------------- WMMA flash attention --------------------------------
__global__ __launch_bounds__(256)
void k_attn_wmma(const bf16_t* __restrict__ Qb, const bf16_t* __restrict__ Kb,
                 const bf16_t* __restrict__ Vt, bf16_t* __restrict__ O) {
  __shared__ __align__(16) __bf16 Pl[8][16][40];
  const int lane   = threadIdx.x & 31;
  const int wib    = threadIdx.x >> 5;
  const int wid    = blockIdx.x * 8 + wib;
  const int h      = wid % cHQ;
  const int qt     = wid / cHQ;
  const int t0     = (qt * 16) % cT;
  const int b      = (qt * 16) / cT;
  const int lane16 = lane & 15;
  const int half   = lane >> 4;

  // Q fragments (A-layout) over hd = 4 x 32
  v16bf qfr[4];
  {
    const bf16_t* qp = Qb + (size_t)(b * cT + t0 + lane16) * (cHQ * cHD) + h * cHD;
#pragma unroll
    for (int c = 0; c < 4; ++c) {
      v8bf q0 = *(const v8bf*)(qp + c * 32 + 8 * half);
      v8bf q1 = *(const v8bf*)(qp + c * 32 + 16 + 8 * half);
#pragma unroll
      for (int e = 0; e < 8; ++e) { qfr[c][e] = q0[e]; qfr[c][8 + e] = q1[e]; }
    }
  }

  v8f oacc[8] = {};
  float mrow[8], lrow[8];
#pragma unroll
  for (int r = 0; r < 8; ++r) { mrow[r] = -3.0e38f; lrow[r] = 0.f; }

  const float scale = 0.08838834764831845f;  // 128^-0.5
  int kstart = t0 - cW; if (kstart < 0) kstart = 0;
  kstart &= ~31;
  const int kend = t0 + 15;

  for (int kc = kstart; kc <= kend; kc += 32) {
    // ---- scores: S[16q x 32k] ----
    v8f s0 = {}, s1 = {};
#pragma unroll
    for (int c = 0; c < 4; ++c) {
      const bf16_t* kp0 = Kb + (size_t)(b * cT + kc + lane16) * cHD + c * 32;
      const bf16_t* kp1 = Kb + (size_t)(b * cT + kc + 16 + lane16) * cHD + c * 32;
      v16bf kf0, kf1;
      v8bf x0 = *(const v8bf*)(kp0 + 8 * half);
      v8bf x1 = *(const v8bf*)(kp0 + 16 + 8 * half);
      v8bf y0 = *(const v8bf*)(kp1 + 8 * half);
      v8bf y1 = *(const v8bf*)(kp1 + 16 + 8 * half);
#pragma unroll
      for (int e = 0; e < 8; ++e) {
        kf0[e] = x0[e]; kf0[8 + e] = x1[e];
        kf1[e] = y0[e]; kf1[8 + e] = y1[e];
      }
      s0 = __builtin_amdgcn_wmma_f32_16x16x32_bf16(false, qfr[c], false, kf0, (short)0, s0, false, false);
      s1 = __builtin_amdgcn_wmma_f32_16x16x32_bf16(false, qfr[c], false, kf1, (short)0, s1, false, false);
    }

    // ---- online softmax over rows ----
    float corr[8];
#pragma unroll
    for (int r = 0; r < 8; ++r) {
      int m = r + 8 * half;           // query row within tile
      int t = t0 + m;
      int j0 = kc + lane16, j1 = kc + 16 + lane16;
      bool v0 = (j0 <= t) && (j0 >= t - cW);
      bool v1 = (j1 <= t) && (j1 >= t - cW);
      float a0 = v0 ? s0[r] * scale : -3.0e38f;
      float a1 = v1 ? s1[r] * scale : -3.0e38f;
      float cand = fmaxf(a0, a1);
#pragma unroll
      for (int mk = 8; mk; mk >>= 1) cand = fmaxf(cand, __shfl_xor(cand, mk, 32));
      float nm = fmaxf(mrow[r], cand);
      float e0 = v0 ? __expf(a0 - nm) : 0.f;
      float e1 = v1 ? __expf(a1 - nm) : 0.f;
      float rs = e0 + e1;
#pragma unroll
      for (int mk = 8; mk; mk >>= 1) rs += __shfl_xor(rs, mk, 32);
      float cr = __expf(mrow[r] - nm);
      lrow[r] = lrow[r] * cr + rs;
      mrow[r] = nm;
      corr[r] = cr;
      Pl[wib][m][lane16]      = (bf16_t)e0;   // transpose P via LDS
      Pl[wib][m][16 + lane16] = (bf16_t)e1;
    }

    // rescale O accumulators
#pragma unroll
    for (int j = 0; j < 8; ++j)
#pragma unroll
      for (int r = 0; r < 8; ++r) oacc[j][r] *= corr[r];

    // reload P in A-fragment layout (16x32)
    v16bf pf;
    {
      const __bf16* pp = &Pl[wib][lane16][0];
      v8bf c0 = *(const v8bf*)(pp + 8 * half);
      v8bf c1 = *(const v8bf*)(pp + 16 + 8 * half);
#pragma unroll
      for (int e = 0; e < 8; ++e) { pf[e] = c0[e]; pf[8 + e] = c1[e]; }
    }

    // ---- O += P * V ---- (B-frag: col = hd, elems = key, from transposed V)
#pragma unroll
    for (int j = 0; j < 8; ++j) {
      const bf16_t* vp = Vt + (size_t)(j * 16 + lane16) * (size_t)cM
                            + (size_t)(b * cT + kc) + 8 * half;
      v16bf vf;
      v8bf w0 = *(const v8bf*)vp;
      v8bf w1 = *(const v8bf*)(vp + 16);
#pragma unroll
      for (int e = 0; e < 8; ++e) { vf[e] = w0[e]; vf[8 + e] = w1[e]; }
      oacc[j] = __builtin_amdgcn_wmma_f32_16x16x32_bf16(false, pf, false, vf, (short)0, oacc[j], false, false);
    }
  }

  // ---- normalize + store ----
#pragma unroll
  for (int r = 0; r < 8; ++r) {
    float inv = 1.f / lrow[r];
    int m = r + 8 * half;
    size_t orow = (size_t)(b * cT + t0 + m) * (cHQ * cHD) + h * cHD;
#pragma unroll
    for (int j = 0; j < 8; ++j)
      O[orow + j * 16 + lane16] = (bf16_t)(oacc[j][r] * inv);
  }
}

// -------------------------- gated-MLP activation ---------------------------
__global__ void k_gelu_gate(const float* __restrict__ G, bf16_t* __restrict__ out,
                            int Hg, long n) {
  long i = (long)blockIdx.x * 256 + threadIdx.x;
  if (i >= n) return;
  long row = i / Hg; int c = (int)(i % Hg);
  float g = G[row * (2 * (long)Hg) + c];
  float h = G[row * (2 * (long)Hg) + Hg + c];
  out[i] = (bf16_t)(gelu_exact(g) * h);
}

// ----------------------- depthwise causal conv (K=4) -----------------------
__global__ void k_conv(const float* __restrict__ GH, const float* __restrict__ cw,
                       const float* __restrict__ cb, float* __restrict__ xcf,
                       bf16_t* __restrict__ xcb) {
  int row = blockIdx.x;
  int t   = row % cT;
  for (int c = threadIdx.x; c < cHH; c += blockDim.x) {
    float acc = cb[c];
#pragma unroll
    for (int j = 0; j < cK; ++j) {
      int tj = t - (cK - 1) + j;
      if (tj >= 0)
        acc += cw[c * cK + j] * GH[(size_t)(row - (cK - 1) + j) * (2 * cHH) + cHH + c];
    }
    xcf[(size_t)row * cHH + c] = acc;
    xcb[(size_t)row * cHH + c] = (bf16_t)acc;
  }
}

// ------------------------ Hawk alpha / beta / xin --------------------------
__global__ void k_alpha_xin(const float* __restrict__ G2, const float* __restrict__ xc,
                            const float* __restrict__ fbase, float* __restrict__ alpha,
                            float* __restrict__ xin, long n) {
  long i = (long)blockIdx.x * 256 + threadIdx.x;
  if (i >= n) return;
  long row = i / cHH; int c = (int)(i % cHH);
  float fg = G2[row * (2 * (long)cHH) + c];
  float ig = G2[row * (2 * (long)cHH) + cHH + c];
  float f  = fbase[c];
  float sp = (f > 20.f) ? f : log1pf(__expf(f));          // softplus
  float sfg = 1.f / (1.f + __expf(-fg));
  float la  = -8.0f * sp * sfg;                            // -exp(log 8)*softplus*sigmoid
  float a   = __expf(la);
  float be  = sqrtf(1.f - a * a + 1e-6f);
  float sig = 1.f / (1.f + __expf(-ig));
  alpha[i] = a;
  xin[i]   = be * sig * xc[i];
}

// ----------------- Hawk recurrence: chunked parallel scan ------------------
// Pass 1: per (b, chunk, c): local scan of h and cumulative alpha product,
//         in-place (xin -> h_local, alpha -> alpha_prefix); chunk summaries.
__global__ void k_scan_local(float* __restrict__ alp, float* __restrict__ xin,
                             float* __restrict__ Asum, float* __restrict__ Xsum) {
  int idx = blockIdx.x * 256 + threadIdx.x;
  if (idx >= cB * kNCH * cHH) return;
  int c  = idx % cHH;
  int ch = (idx / cHH) % kNCH;
  int b  = idx / (cHH * kNCH);
  float h = 0.f, pa = 1.f;
  size_t base = ((size_t)b * cT + (size_t)ch * kSC) * cHH + c;
  for (int t = 0; t < kSC; ++t) {
    size_t o = base + (size_t)t * cHH;
    if (t + 8 < kSC) {   // stream-prefetch the dependent chain
      __builtin_prefetch(&alp[o + 8 * (size_t)cHH], 0, 0);
      __builtin_prefetch(&xin[o + 8 * (size_t)cHH], 0, 0);
    }
    float a = alp[o];
    h  = a * h + xin[o];
    pa = pa * a;
    xin[o] = h;
    alp[o] = pa;
  }
  Asum[idx] = pa;
  Xsum[idx] = h;
}

// Pass 2: per (b, c): serial scan over the 16 chunk summaries -> carry-ins.
__global__ void k_scan_carry(const float* __restrict__ Asum, const float* __restrict__ Xsum,
                             float* __restrict__ Carry) {
  int idx = blockIdx.x * 256 + threadIdx.x;
  if (idx >= cB * cHH) return;
  int c = idx % cHH, b = idx / cHH;
  float carry = 0.f;
  for (int ch = 0; ch < kNCH; ++ch) {
    size_t o = ((size_t)b * kNCH + ch) * cHH + c;
    Carry[o] = carry;
    carry = Asum[o] * carry + Xsum[o];
  }
}

// Pass 3: fully parallel fixup: h += carry(chunk) * alpha_prefix.
__global__ void k_scan_apply(const float* __restrict__ alpPre, float* __restrict__ h,
                             const float* __restrict__ Carry, long n) {
  long i = (long)blockIdx.x * 256 + threadIdx.x;
  if (i >= n) return;
  int c   = (int)(i % cHH);
  long bt = i / cHH;
  int t = (int)(bt % cT);
  int b = (int)(bt / cT);
  int ch = t / kSC;
  float carry = Carry[((size_t)b * kNCH + ch) * cHH + c];
  h[i] += carry * alpPre[i];
}

// out bf16 [M, HH] = gelu(GH.gate) * H
__global__ void k_hawk_act(const float* __restrict__ GH, const float* __restrict__ H,
                           bf16_t* __restrict__ out, long n) {
  long i = (long)blockIdx.x * 256 + threadIdx.x;
  if (i >= n) return;
  long row = i / cHH; int c = (int)(i % cHH);
  float g = GH[row * (2 * (long)cHH) + c];
  out[i] = (bf16_t)(gelu_exact(g) * H[i]);
}

// ------------------------------- driver ------------------------------------

static inline dim3 gemm_grid(int M, int N) { return dim3(N / 128, M / 64, 1); }

extern "C" void kernel_launch(void* const* d_in, const int* in_sizes, int n_in,
                              void* d_out, int out_size, void* d_ws, size_t ws_size,
                              hipStream_t stream) {
  (void)in_sizes; (void)n_in; (void)out_size; (void)ws_size;

  const float* x         = (const float*)d_in[0];
  const float* sn_g      = (const float*)d_in[1];
  const float* q_w       = (const float*)d_in[2];
  const float* kv_w      = (const float*)d_in[3];
  const float* ao_w      = (const float*)d_in[4];
  const float* sgn_g     = (const float*)d_in[5];
  const float* sg_grow   = (const float*)d_in[6];
  const float* sg_shrink = (const float*)d_in[7];
  const float* hn_g      = (const float*)d_in[8];
  const float* h_in_w    = (const float*)d_in[9];
  const float* h_conv_w  = (const float*)d_in[10];
  const float* h_conv_b  = (const float*)d_in[11];
  const float* h_gates_w = (const float*)d_in[12];
  const float* h_gates_b = (const float*)d_in[13];
  const float* h_forget  = (const float*)d_in[14];
  const float* h_out_w   = (const float*)d_in[15];
  const float* hgn_g     = (const float*)d_in[16];
  const float* hg_grow   = (const float*)d_in[17];
  const float* hg_shrink = (const float*)d_in[18];
  float* out = (float*)d_out;

  char* p = (char*)d_ws;
  auto carve = [&](size_t bytes) -> char* {
    char* r = p; p += (bytes + 255) & ~(size_t)255; return r;
  };

  // bf16 weight pool
  bf16_t* WQ   = (bf16_t*)carve((size_t)cD * cD * 2);
  bf16_t* WKV  = (bf16_t*)carve((size_t)256 * cD * 2);
  bf16_t* WAO  = (bf16_t*)carve((size_t)cD * cD * 2);
  bf16_t* WSGG = (bf16_t*)carve((size_t)2 * cHG * cD * 2);
  bf16_t* WSGS = (bf16_t*)carve((size_t)cD * cHG * 2);
  bf16_t* WHIN = (bf16_t*)carve((size_t)2 * cHH * cD * 2);
  bf16_t* WHGA = (bf16_t*)carve((size_t)2 * cHH * cHH * 2);
  bf16_t* WHO  = (bf16_t*)carve((size_t)cD * cHH * 2);
  bf16_t* WHGG = (bf16_t*)carve((size_t)2 * cHG * cD * 2);
  bf16_t* WHGS = (bf16_t*)carve((size_t)cD * cHG * 2);

  // activations
  bf16_t* XBF  = (bf16_t*)carve((size_t)cM * cD * 2);           // normed x
  bf16_t* ACT  = (bf16_t*)carve((size_t)cM * 2 * cHG * 2);      // attn-O / MLP / hawk acts
  bf16_t* XCB  = (bf16_t*)carve((size_t)cM * cHH * 2);          // conv out bf16
  bf16_t* QBF  = (bf16_t*)carve((size_t)cM * cD * 2);           // roped Q bf16
  bf16_t* KBF  = (bf16_t*)carve((size_t)cM * cHD * 2);          // roped K bf16
  bf16_t* VTB  = (bf16_t*)carve((size_t)cHD * cM * 2);          // V transposed bf16
  float*  G1   = (float*)carve((size_t)cM * 2 * cHG * 4);       // 8192x4096 f32
  float*  G2   = (float*)carve((size_t)cM * 2 * cHH * 4);       // 8192x3072 f32
  float*  Qf   = (float*)carve((size_t)cM * cHQ * cHD * 4);
  float*  KVf  = (float*)carve((size_t)cM * 256 * 4);
  float*  XCf  = (float*)carve((size_t)cM * cHH * 4);
  float*  ALP  = (float*)carve((size_t)cM * cHH * 4);           // becomes alpha-prefix
  float*  XIN  = (float*)carve((size_t)cM * cHH * 4);           // becomes H
  float*  ASUM = (float*)carve((size_t)cB * kNCH * cHH * 4);
  float*  XSUM = (float*)carve((size_t)cB * kNCH * cHH * 4);
  float*  CARY = (float*)carve((size_t)cB * kNCH * cHH * 4);

  auto cvt = [&](const float* s, bf16_t* d, int n) {
    k_f32_to_bf16<<<(n + 255) / 256, 256, 0, stream>>>(s, d, n);
  };
  cvt(q_w,       WQ,   cD * cD);
  cvt(kv_w,      WKV,  256 * cD);
  cvt(ao_w,      WAO,  cD * cD);
  cvt(sg_grow,   WSGG, 2 * cHG * cD);
  cvt(sg_shrink, WSGS, cD * cHG);
  cvt(h_in_w,    WHIN, 2 * cHH * cD);
  cvt(h_gates_w, WHGA, 2 * cHH * cHH);
  cvt(h_out_w,   WHO,  cD * cHH);
  cvt(hg_grow,   WHGG, 2 * cHG * cD);
  cvt(hg_shrink, WHGS, cD * cHG);

  // residual stream lives in d_out
  k_copy_f32<<<(cM * cD + 255) / 256, 256, 0, stream>>>(x, out, cM * cD);

  // ---- 1) sliding-window MQA ----
  k_rmsnorm_bf16<<<cM, 256, 0, stream>>>(out, sn_g, XBF, cD);
  k_gemm_bf16_wmma<0><<<gemm_grid(cM, cD), 256, 0, stream>>>(XBF, WQ, Qf, nullptr, cM, cD, cD);
  k_gemm_bf16_wmma<0><<<gemm_grid(cM, 256), 256, 0, stream>>>(XBF, WKV, KVf, nullptr, cM, 256, cD);
  k_rope<<<cM, 256, 0, stream>>>(Qf, KVf, QBF, KBF, VTB);
  k_attn_wmma<<<(cM / 16) * cHQ / 8, 256, 0, stream>>>(QBF, KBF, VTB, ACT);
  k_gemm_bf16_wmma<1><<<gemm_grid(cM, cD), 256, 0, stream>>>(ACT, WAO, out, nullptr, cM, cD, cD);

  // ---- 2) gated MLP ----
  k_rmsnorm_bf16<<<cM, 256, 0, stream>>>(out, sgn_g, XBF, cD);
  k_gemm_bf16_wmma<0><<<gemm_grid(cM, 2 * cHG), 256, 0, stream>>>(XBF, WSGG, G1, nullptr, cM, 2 * cHG, cD);
  {
    long n = (long)cM * cHG;
    k_gelu_gate<<<(int)((n + 255) / 256), 256, 0, stream>>>(G1, ACT, cHG, n);
  }
  k_gemm_bf16_wmma<1><<<gemm_grid(cM, cD), 256, 0, stream>>>(ACT, WSGS, out, nullptr, cM, cD, cHG);

  // ---- 3) Hawk ----
  k_rmsnorm_bf16<<<cM, 256, 0, stream>>>(out, hn_g, XBF, cD);
  k_gemm_bf16_wmma<0><<<gemm_grid(cM, 2 * cHH), 256, 0, stream>>>(XBF, WHIN, G1, nullptr, cM, 2 * cHH, cD);
  k_conv<<<cM, 256, 0, stream>>>(G1, h_conv_w, h_conv_b, XCf, XCB);
  k_gemm_bf16_wmma<0><<<gemm_grid(cM, 2 * cHH), 256, 0, stream>>>(XCB, WHGA, G2, h_gates_b, cM, 2 * cHH, cHH);
  {
    long n = (long)cM * cHH;
    k_alpha_xin<<<(int)((n + 255) / 256), 256, 0, stream>>>(G2, XCf, h_forget, ALP, XIN, n);
  }
  {
    int nloc = cB * kNCH * cHH;
    k_scan_local<<<(nloc + 255) / 256, 256, 0, stream>>>(ALP, XIN, ASUM, XSUM);
    k_scan_carry<<<(cB * cHH + 255) / 256, 256, 0, stream>>>(ASUM, XSUM, CARY);
    long n = (long)cM * cHH;
    k_scan_apply<<<(int)((n + 255) / 256), 256, 0, stream>>>(ALP, XIN, CARY, n);
  }
  {
    long n = (long)cM * cHH;
    k_hawk_act<<<(int)((n + 255) / 256), 256, 0, stream>>>(G1, XIN, ACT, n);
  }
  k_gemm_bf16_wmma<1><<<gemm_grid(cM, cD), 256, 0, stream>>>(ACT, WHO, out, nullptr, cM, cD, cHH);

  // ---- 4) gated MLP ----
  k_rmsnorm_bf16<<<cM, 256, 0, stream>>>(out, hgn_g, XBF, cD);
  k_gemm_bf16_wmma<0><<<gemm_grid(cM, 2 * cHG), 256, 0, stream>>>(XBF, WHGG, G1, nullptr, cM, 2 * cHG, cD);
  {
    long n = (long)cM * cHG;
    k_gelu_gate<<<(int)((n + 255) / 256), 256, 0, stream>>>(G1, ACT, cHG, n);
  }
  k_gemm_bf16_wmma<1><<<gemm_grid(cM, cD), 256, 0, stream>>>(ACT, WHGS, out, nullptr, cM, cD, cHG);
}
